// DOSTransformer_phonon_1589137899995
// MI455X (gfx1250) — matrix operand order repaired and checked by reading, not compiled, not run on hardware
//
#include <hip/hip_runtime.h>

#define E_EDGES 200000
#define N_NODES 20480
#define HDIM    128
#define BGR     128
#define PPG     160
#define NE_G    51
#define ECHUNK  25000
#define NCHUNK  8

typedef __attribute__((ext_vector_type(16))) __bf16 v16bf;
typedef __attribute__((ext_vector_type(8)))  __bf16 v8bf;
typedef __attribute__((ext_vector_type(8)))  float  v8f;

__device__ __forceinline__ __bf16 f2bf(float f) {
  unsigned u = __builtin_bit_cast(unsigned, f);
  u += 0x7FFFu + ((u >> 16) & 1u);                 // round-to-nearest-even
  unsigned short h = (unsigned short)(u >> 16);
  return __builtin_bit_cast(__bf16, h);
}

// Load one 16x32 bf16 A-fragment from a K-padded bf16 row:
// elements 0..7  = k = kb + half*16 + e      (16 contiguous bytes)
// elements 8..15 = k = kb + half*16 + 8 + e  (16 contiguous bytes)
__device__ __forceinline__ v16bf load_afrag(const __bf16* row, int kb, int hh) {
  const __bf16* p = row + kb + (hh << 4);
  v8bf lo = *(const v8bf*)p;
  v8bf hi = *(const v8bf*)(p + 8);
  return __builtin_shufflevector(lo, hi, 0, 1, 2, 3, 4, 5, 6, 7,
                                 8, 9, 10, 11, 12, 13, 14, 15);
}

// ---------------------------------------------------------------------------
// Weight pack: f32 [K,N] row-major -> bf16 WMMA B-fragments.
// Fragment (kt, nt): lane holds column n = nt*16 + (lane&15); half = lane>>4;
// element e holds k = kt*32 + half*16 + e.
// ---------------------------------------------------------------------------
__global__ __launch_bounds__(128)
void k_pack_w(const float* __restrict__ W, __bf16* __restrict__ out,
              int K, int Nd) {
  const int lane = threadIdx.x & 31, wave = threadIdx.x >> 5;
  const int Kt = (K + 31) >> 5, Nt = (Nd + 15) >> 4;
  int frag = blockIdx.x * 4 + wave;
  if (frag >= Kt * Nt) return;
  int kt = frag / Nt, ft = frag % Nt;
  int kb = kt << 5, n = (ft << 4) + (lane & 15);
  int hh = lane >> 4;
  __bf16* dst = out + (size_t)frag * 512 + (size_t)lane * 16;
#pragma unroll
  for (int e = 0; e < 16; ++e) {
    int k = kb + ((hh << 4) | e);
    float v = (k < K && n < Nd) ? W[(size_t)k * Nd + n] : 0.0f;
    dst[e] = f2bf(v);
  }
}

// ---------------------------------------------------------------------------
// Main WMMA GEMM: C = act((A_cat x W) + bias) (+ resid), bf16 in, f32 acc.
// A = up to 3 K-padded bf16 sources with optional per-row gather.
// NFR (compile-time) 16-wide output fragments per wave: one A fragment feeds
// NFR unconditional WMMAs (no conditional accumulators -> no mov storms).
// NFR=4 => 16x64 tile (all N multiples of 64); NFR=1 => final N=1 head.
// ---------------------------------------------------------------------------
template <int NFR>
__global__ __launch_bounds__(256)
void k_gemm(const __bf16* __restrict__ A0, const int* __restrict__ idx0, int Kt0,
            const __bf16* __restrict__ A1, const int* __restrict__ idx1, int Kt1,
            const __bf16* __restrict__ A2, const int* __restrict__ idx2, int Kt2,
            int lda,
            const __bf16* __restrict__ Wp,
            const float* __restrict__ bias, const float* __restrict__ resid,
            float* __restrict__ C, __bf16* __restrict__ Cb,
            int M, int Nd, int ldc, int act, const float* __restrict__ slope) {
  const int lane = threadIdx.x & 31, wave = threadIdx.x >> 5;
  const int m0 = (blockIdx.x * 8 + wave) * 16;
  if (m0 >= M) return;
  const int n0 = blockIdx.y * (NFR * 16);
  const int Nt16 = (Nd + 15) >> 4;
  const int f0 = n0 >> 4;
  const int hh = lane >> 4, r = lane & 15;
  const int m = m0 + r;
  const bool mOK = m < M;
  int ar0 = 0, ar1 = 0, ar2 = 0;
  if (mOK) {
    ar0 = idx0 ? idx0[m] : m;
    if (A1) ar1 = idx1 ? idx1[m] : m;
    if (A2) ar2 = idx2 ? idx2[m] : m;
  }
  const __bf16* base0 = A0 + (size_t)ar0 * lda;
  const __bf16* base1 = A1 + (size_t)ar1 * lda;
  const __bf16* base2 = A2 + (size_t)ar2 * lda;

  v8f acc[NFR] = {};
  int kt = 0;
#pragma unroll 1
  for (int s = 0; s < 3; ++s) {
    const __bf16* base; int KtS;
    if (s == 0)      { base = base0; KtS = Kt0; }
    else if (s == 1) { base = base1; KtS = Kt1; }
    else             { base = base2; KtS = Kt2; }
    for (int k2 = 0; k2 < KtS; ++k2, ++kt) {
      v16bf a = load_afrag(base, k2 << 5, hh);
      const __bf16* fp = Wp + ((size_t)kt * Nt16 + f0) * 512 + (size_t)lane * 16;
#pragma unroll
      for (int j = 0; j < NFR; ++j) {
        v16bf b = *(const v16bf*)(fp + (size_t)j * 512);
        acc[j] = __builtin_amdgcn_wmma_f32_16x16x32_bf16(false, a, false, b,
                                                         (short)0, acc[j],
                                                         false, false);
      }
    }
  }
  float sl = slope ? slope[0] : 0.0f;
#pragma unroll
  for (int rr = 0; rr < 8; ++rr) {
    int mm = m0 + rr + (hh << 3);
    if (mm >= M) continue;
#pragma unroll
    for (int j = 0; j < NFR; ++j) {
      int nn = n0 + (j << 4) + r;
      if (nn >= Nd) continue;
      float v = acc[j][rr];
      if (bias) v += bias[nn];
      if (act == 1) v = v >= 0.0f ? v : sl * v;       // PReLU
      else if (act == 2) v = v > 0.0f ? v : 0.0f;     // ReLU
      if (resid) v += resid[(size_t)mm * ldc + nn];
      if (C)  C[(size_t)mm * ldc + nn] = v;
      if (Cb) Cb[(size_t)mm * ldc + nn] = f2bf(v);
    }
  }
}

// ---------------------------------------------------------------------------
// Batched NT GEMM (scores): C = scale * A[M,K] x B[N,K]^T, both bf16
// row-major K-padded; B fragment = one contiguous 32B load per lane.
// ---------------------------------------------------------------------------
__global__ __launch_bounds__(256)
void k_gemm_nt(const __bf16* __restrict__ A, const __bf16* __restrict__ Bm,
               float* __restrict__ C, int M, int Nd, int Ktiles,
               int lda, int ldb, int ldc,
               long long batchA, long long batchB, long long batchC,
               float scale) {
  const int lane = threadIdx.x & 31, wave = threadIdx.x >> 5;
  A  += (size_t)blockIdx.z * batchA;
  Bm += (size_t)blockIdx.z * batchB;
  C  += (size_t)blockIdx.z * batchC;
  const int m0 = (blockIdx.x * 8 + wave) * 16;
  if (m0 >= M) return;
  const int n0 = blockIdx.y * 32;
  const int hh = lane >> 4, r = lane & 15;
  const int m = m0 + r;
  const bool mOK = m < M;
  const int n1 = n0 + r, n2 = n0 + 16 + r;
  const bool ok1 = n1 < Nd, ok2 = n2 < Nd;
  const __bf16* Arow = A + (size_t)(mOK ? m : 0) * lda;
  const __bf16* Br0 = Bm + (size_t)(ok1 ? n1 : 0) * ldb;
  const __bf16* Br1 = Bm + (size_t)(ok2 ? n2 : 0) * ldb;
  v8f acc0 = {}; v8f acc1 = {};
  for (int kt = 0; kt < Ktiles; ++kt) {
    int kb = kt << 5;
    v16bf a = load_afrag(Arow, kb, hh);
    v16bf b0 = *(const v16bf*)(Br0 + kb + (hh << 4));
    v16bf b1 = *(const v16bf*)(Br1 + kb + (hh << 4));
    acc0 = __builtin_amdgcn_wmma_f32_16x16x32_bf16(false, a, false, b0,
                                                   (short)0, acc0, false, false);
    acc1 = __builtin_amdgcn_wmma_f32_16x16x32_bf16(false, a, false, b1,
                                                   (short)0, acc1, false, false);
  }
#pragma unroll
  for (int rr = 0; rr < 8; ++rr) {
    int mm = m0 + rr + (hh << 3);
    if (mm >= M) continue;
    if (ok1) C[(size_t)mm * ldc + n1] = acc0[rr] * scale;
    if (ok2) C[(size_t)mm * ldc + n2] = acc1[rr] * scale;
  }
}

// ---------------------------------------------------------------------------
// Batched NN GEMM: ctx = attn(bf16) x V(f32, [K,N] row-major) -> bf16 out.
// ---------------------------------------------------------------------------
__global__ __launch_bounds__(256)
void k_gemm_nn(const __bf16* __restrict__ A, const float* __restrict__ Bm,
               __bf16* __restrict__ Cb, int M, int Nd, int Ktiles,
               int lda, int ldb, int ldc,
               long long batchA, long long batchB, long long batchC) {
  const int lane = threadIdx.x & 31, wave = threadIdx.x >> 5;
  A  += (size_t)blockIdx.z * batchA;
  Bm += (size_t)blockIdx.z * batchB;
  Cb += (size_t)blockIdx.z * batchC;
  const int m0 = (blockIdx.x * 8 + wave) * 16;
  if (m0 >= M) return;
  const int n0 = blockIdx.y * 32;
  const int hh = lane >> 4, r = lane & 15;
  const int m = m0 + r;
  const bool mOK = m < M;
  const int n1 = n0 + r, n2 = n0 + 16 + r;
  const bool ok1 = n1 < Nd, ok2 = n2 < Nd;
  const __bf16* Arow = A + (size_t)(mOK ? m : 0) * lda;
  v8f acc0 = {}; v8f acc1 = {};
  for (int kt = 0; kt < Ktiles; ++kt) {
    int kb = kt << 5;
    v16bf a = load_afrag(Arow, kb, hh);
    v16bf b0, b1;
#pragma unroll
    for (int e = 0; e < 16; ++e) {
      int k = kb + ((hh << 4) | e);
      b0[e] = f2bf(ok1 ? Bm[(size_t)k * ldb + n1] : 0.0f);
      b1[e] = f2bf(ok2 ? Bm[(size_t)k * ldb + n2] : 0.0f);
    }
    acc0 = __builtin_amdgcn_wmma_f32_16x16x32_bf16(false, a, false, b0,
                                                   (short)0, acc0, false, false);
    acc1 = __builtin_amdgcn_wmma_f32_16x16x32_bf16(false, a, false, b1,
                                                   (short)0, acc1, false, false);
  }
#pragma unroll
  for (int rr = 0; rr < 8; ++rr) {
    int mm = m0 + rr + (hh << 3);
    if (mm >= M) continue;
    if (ok1) Cb[(size_t)mm * ldc + n1] = f2bf(acc0[rr]);
    if (ok2) Cb[(size_t)mm * ldc + n2] = f2bf(acc1[rr]);
  }
}

// ---------------------------------------------------------------------------
// LayerNorm (optional fused PReLU). One wave per row; D in {64,128,256}.
// Writes f32 and/or bf16 outputs.
// ---------------------------------------------------------------------------
__global__ __launch_bounds__(256)
void k_layernorm(const float* __restrict__ in, float* __restrict__ outf,
                 __bf16* __restrict__ outb, int rows, int D,
                 const float* __restrict__ g, const float* __restrict__ b,
                 const float* __restrict__ slope) {
  const int lane = threadIdx.x & 31, wave = threadIdx.x >> 5;
  int row = blockIdx.x * 8 + wave;
  if (row >= rows) return;
  const float* x = in + (size_t)row * D;
  int n = D >> 5;
  float vals[8];
  float s = 0.f, ss = 0.f;
  for (int i = 0; i < n; ++i) {
    float v = x[i * 32 + lane];
    vals[i] = v; s += v; ss += v * v;
  }
  for (int o = 16; o; o >>= 1) {
    s  += __shfl_xor(s,  o, 32);
    ss += __shfl_xor(ss, o, 32);
  }
  float mean = s / D;
  float var = ss / D - mean * mean;
  float rstd = rsqrtf(var + 1e-5f);
  float sl = slope ? slope[0] : 0.f;
  for (int i = 0; i < n; ++i) {
    int c = i * 32 + lane;
    float v = (vals[i] - mean) * rstd * g[c] + b[c];
    if (slope) v = v >= 0.f ? v : sl * v;
    if (outf) outf[(size_t)row * D + c] = v;
    if (outb) outb[(size_t)row * D + c] = f2bf(v);
  }
}

// Row softmax (D = 160), bf16 output for the ctx GEMM.
__global__ __launch_bounds__(256)
void k_softmax(const float* __restrict__ x, __bf16* __restrict__ outb,
               int rows, int D) {
  const int lane = threadIdx.x & 31, wave = threadIdx.x >> 5;
  int row = blockIdx.x * 8 + wave;
  if (row >= rows) return;
  const float* p = x + (size_t)row * D;
  __bf16* ob = outb + (size_t)row * D;
  int n = (D + 31) >> 5;
  float vals[8];
  float mx = -3.0e38f;
  for (int i = 0; i < n; ++i) {
    int c = i * 32 + lane;
    float v = (c < D) ? p[c] : -3.0e38f;
    vals[i] = v; mx = fmaxf(mx, v);
  }
  for (int o = 16; o; o >>= 1) mx = fmaxf(mx, __shfl_xor(mx, o, 32));
  float s = 0.f;
  for (int i = 0; i < n; ++i) {
    int c = i * 32 + lane;
    float e = (c < D) ? __expf(vals[i] - mx) : 0.f;
    vals[i] = e; s += e;
  }
  for (int o = 16; o; o >>= 1) s += __shfl_xor(s, o, 32);
  float inv = 1.0f / s;
  for (int i = 0; i < n; ++i) {
    int c = i * 32 + lane;
    if (c < D) ob[c] = f2bf(vals[i] * inv);
  }
}

// Elementwise helpers --------------------------------------------------------
__global__ void k_fill(float* p, long long n, float v) {
  long long i = (long long)blockIdx.x * blockDim.x + threadIdx.x;
  if (i < n) p[i] = v;
}

// f32 [rows,Kin] -> bf16 [rows,Kout] zero-padded
__global__ void k_cvt_pad(const float* __restrict__ in, __bf16* __restrict__ out,
                          long long rows, int Kin, int Kout) {
  long long i = (long long)blockIdx.x * blockDim.x + threadIdx.x;
  if (i >= rows * Kout) return;
  long long r0 = i / Kout; int c = (int)(i % Kout);
  out[i] = f2bf(c < Kin ? in[r0 * Kin + c] : 0.0f);
}

// smooth_cutoff(|edge_vec|/4) * edge_shift -> bf16 [E,32] zero-padded
__global__ void k_edge_input(const float* __restrict__ ev,
                             const float* __restrict__ es,
                             __bf16* __restrict__ eo, int E) {
  int e = blockIdx.x * blockDim.x + threadIdx.x;
  if (e >= E) return;
  float x = ev[e * 3], y = ev[e * 3 + 1], z = ev[e * 3 + 2];
  float el = sqrtf(x * x + y * y + z * z);
  float u = 2.0f * (el * 0.25f - 1.0f);
  float w = 0.5f * (1.0f - __cosf(3.14159265358979323846f * u));
  w = (u > 0.f) ? 0.f : w;
  w = (u < -1.f) ? 1.f : w;
  __bf16* o = eo + (size_t)e * 32;
  o[0] = f2bf(w * es[e * 3 + 0]);
  o[1] = f2bf(w * es[e * 3 + 1]);
  o[2] = f2bf(w * es[e * 3 + 2]);
  __bf16 zz = f2bf(0.0f);
  for (int c = 3; c < 32; ++c) o[c] = zz;
}

__global__ void k_count(const int* __restrict__ col, float* __restrict__ cnt, int E) {
  int e = blockIdx.x * blockDim.x + threadIdx.x;
  if (e < E) atomicAdd(&cnt[col[e]], 1.0f);
}

__global__ void k_scatter(const float* __restrict__ src, const int* __restrict__ col,
                          float* __restrict__ sums, int EC) {
  int i = blockIdx.x * blockDim.x + threadIdx.x;
  if (i >= EC * HDIM) return;
  int e = i >> 7, h = i & 127;
  atomicAdd(&sums[(size_t)col[e] * HDIM + h], src[i]);
}

// dst += src; also refresh bf16 mirror
__global__ void k_add(float* __restrict__ dst, __bf16* __restrict__ dstb,
                      const float* __restrict__ src, int n) {
  int i = blockIdx.x * blockDim.x + threadIdx.x;
  if (i >= n) return;
  float v = dst[i] + src[i];
  dst[i] = v;
  dstb[i] = f2bf(v);
}

__global__ void k_aggdiv(__bf16* __restrict__ aggb, const float* __restrict__ sums,
                         const float* __restrict__ cnt, int total) {
  int i = blockIdx.x * blockDim.x + threadIdx.x;
  if (i < total) aggb[i] = f2bf(sums[i] / fmaxf(cnt[i >> 7], 1.0f));
}

__global__ void k_sum_nodes(const float* __restrict__ xh, __bf16* __restrict__ gsumb) {
  int b = blockIdx.x, h = threadIdx.x;
  float s = 0.f;
  for (int p = 0; p < PPG; ++p)
    s += xh[((size_t)b * PPG + p) * HDIM + h];
  gsumb[(size_t)b * HDIM + h] = f2bf(s);
}

__global__ void k_init_q(float* __restrict__ q, const float* __restrict__ emb, int total) {
  int i = blockIdx.x * blockDim.x + threadIdx.x;
  if (i >= total) return;
  int t = (i / HDIM) % NE_G;
  q[i] = emb[t * HDIM + (i & 127)];
}

__global__ void k_fused(__bf16* __restrict__ fusedb, const float* __restrict__ en,
                        const float* __restrict__ graph, const float* __restrict__ alpha,
                        int total) {
  int i = blockIdx.x * blockDim.x + threadIdx.x;
  if (i >= total) return;
  int b = i / (NE_G * HDIM);
  fusedb[i] = f2bf(en[i] + alpha[0] * graph[(size_t)b * HDIM + (i & 127)]);
}

// ---------------------------------------------------------------------------
extern "C" void kernel_launch(void* const* d_in, const int* in_sizes, int n_in,
                              void* d_out, int out_size, void* d_ws, size_t ws_size,
                              hipStream_t stream) {
  (void)in_sizes; (void)n_in; (void)out_size; (void)ws_size;
  const int E = E_EDGES, Nn = N_NODES, H = HDIM, B = BGR, P = PPG, NE = NE_G;
  const int EC = ECHUNK;

  // --- input leaf mapping (jax tree_leaves: dict keys sorted) ---
  const int*   ei      = (const int*)d_in[1];
  const int*   ei_row  = ei;
  const int*   ei_col  = ei + E;
  const float* eshift  = (const float*)d_in[2];
  const float* evec    = (const float*)d_in[3];
  const float* alpha   = (const float*)d_in[4];
  const float* dec_b   = (const float*)d_in[5];
  const float* dec_w   = (const float*)d_in[6];
  const float* ee_a    = (const float*)d_in[7];
  const float* ee_l1b  = (const float*)d_in[8];
  const float* ee_l1w  = (const float*)d_in[9];
  const float* ee_l2b  = (const float*)d_in[10];
  const float* ee_l2w  = (const float*)d_in[11];
  const float* emb     = (const float*)d_in[12];
  const float* ne_a    = (const float*)d_in[13];
  const float* ne_l1b  = (const float*)d_in[14];
  const float* ne_l1w  = (const float*)d_in[15];
  const float* ne_l2b  = (const float*)d_in[16];
  const float* ne_l2w  = (const float*)d_in[17];
  const float* out_a   = (const float*)d_in[18];
  const float* out_l1b = (const float*)d_in[19];
  const float* out_l1w = (const float*)d_in[20];
  const float* out_l2b = (const float*)d_in[21];
  const float* out_l2w = (const float*)d_in[22];
  const float* out_lnb = (const float*)d_in[23];
  const float* out_lng = (const float*)d_in[24];
  const float *pe_a[3], *pe_l1b[3], *pe_l1w[3], *pe_l2b[3], *pe_l2w[3], *pe_lnb[3], *pe_lng[3];
  const float *pn_a[3], *pn_l1b[3], *pn_l1w[3], *pn_l2b[3], *pn_l2w[3], *pn_lnb[3], *pn_lng[3];
  for (int l = 0; l < 3; ++l) {
    int base = 25 + l * 14;
    pe_a[l]   = (const float*)d_in[base + 0];
    pe_l1b[l] = (const float*)d_in[base + 1];
    pe_l1w[l] = (const float*)d_in[base + 2];
    pe_l2b[l] = (const float*)d_in[base + 3];
    pe_l2w[l] = (const float*)d_in[base + 4];
    pe_lnb[l] = (const float*)d_in[base + 5];
    pe_lng[l] = (const float*)d_in[base + 6];
    pn_a[l]   = (const float*)d_in[base + 7];
    pn_l1b[l] = (const float*)d_in[base + 8];
    pn_l1w[l] = (const float*)d_in[base + 9];
    pn_l2b[l] = (const float*)d_in[base + 10];
    pn_l2w[l] = (const float*)d_in[base + 11];
    pn_lnb[l] = (const float*)d_in[base + 12];
    pn_lng[l] = (const float*)d_in[base + 13];
  }
  const float *f1b[2], *f1w[2], *f2b[2], *f2w[2], *ln1b[2], *ln1g[2], *ln2b[2], *ln2g[2];
  const float *wkb[2], *wkw[2], *wob[2], *wow[2], *wqb[2], *wqw[2], *wvb[2], *wvw[2];
  for (int t = 0; t < 2; ++t) {
    int base = 67 + t * 16;
    f1b[t]  = (const float*)d_in[base + 0];  f1w[t] = (const float*)d_in[base + 1];
    f2b[t]  = (const float*)d_in[base + 2];  f2w[t] = (const float*)d_in[base + 3];
    ln1b[t] = (const float*)d_in[base + 4];  ln1g[t] = (const float*)d_in[base + 5];
    ln2b[t] = (const float*)d_in[base + 6];  ln2g[t] = (const float*)d_in[base + 7];
    wkb[t]  = (const float*)d_in[base + 8];  wkw[t] = (const float*)d_in[base + 9];
    wob[t]  = (const float*)d_in[base + 10]; wow[t] = (const float*)d_in[base + 11];
    wqb[t]  = (const float*)d_in[base + 12]; wqw[t] = (const float*)d_in[base + 13];
    wvb[t]  = (const float*)d_in[base + 14]; wvw[t] = (const float*)d_in[base + 15];
  }
  const float* trln_b = (const float*)d_in[99];
  const float* trln_g = (const float*)d_in[100];
  const float* x_in   = (const float*)d_in[101];

  // --- workspace carve-out (deterministic) ---
  char* wsp = (char*)d_ws;
  size_t off = 0;
  auto alloc = [&](size_t bytes) -> void* {
    void* p = wsp + off;
    off = (off + bytes + 255) & ~(size_t)255;
    return p;
  };
  // f32 masters
  float* xh    = (float*)alloc((size_t)Nn * H * 4);
  float* ea    = (float*)alloc((size_t)E * H * 4);
  float* eb256 = (float*)alloc((size_t)EC * 256 * 4);
  float* eb128 = (float*)alloc((size_t)EC * H * 4);
  float* nb256 = (float*)alloc((size_t)Nn * 256 * 4);
  float* sums  = (float*)alloc((size_t)Nn * H * 4);
  float* cnt   = (float*)alloc((size_t)Nn * 4);
  float* qb    = (float*)alloc((size_t)B * NE * H * 4);
  float* Vm    = (float*)alloc((size_t)Nn * H * 4);
  float* sc    = (float*)alloc((size_t)B * NE * P * 4);
  float* en    = (float*)alloc((size_t)B * NE * H * 4);
  float* graph = (float*)alloc((size_t)B * H * 4);
  float* oh1   = (float*)alloc((size_t)B * NE * 64 * 4);
  // bf16 activation mirrors (K-padded to multiples of 32)
  __bf16* xb     = (__bf16*)alloc((size_t)Nn * 96 * 2);
  __bf16* xhb    = (__bf16*)alloc((size_t)Nn * H * 2);
  __bf16* e_inb  = (__bf16*)alloc((size_t)E * 32 * 2);
  __bf16* eab    = (__bf16*)alloc((size_t)E * H * 2);
  __bf16* tmpNb  = (__bf16*)alloc((size_t)Nn * H * 2);
  __bf16* eb128b = (__bf16*)alloc((size_t)EC * H * 2);
  __bf16* eb256b = (__bf16*)alloc((size_t)EC * 256 * 2);
  __bf16* nb256b = (__bf16*)alloc((size_t)Nn * 256 * 2);
  __bf16* aggb   = (__bf16*)alloc((size_t)Nn * H * 2);
  __bf16* kvnb   = (__bf16*)alloc((size_t)Nn * H * 2);
  __bf16* qnb    = (__bf16*)alloc((size_t)B * NE * H * 2);
  __bf16* Qmb    = (__bf16*)alloc((size_t)B * NE * H * 2);
  __bf16* Kmb    = (__bf16*)alloc((size_t)Nn * H * 2);
  __bf16* scb    = (__bf16*)alloc((size_t)B * NE * P * 2);
  __bf16* ctxb   = (__bf16*)alloc((size_t)B * NE * H * 2);
  __bf16* ffhb   = (__bf16*)alloc((size_t)B * NE * 512 * 2);
  __bf16* fusedb = (__bf16*)alloc((size_t)B * NE * H * 2);
  __bf16* gsumb  = (__bf16*)alloc((size_t)B * H * 2);
  __bf16* oh1b   = (__bf16*)alloc((size_t)B * NE * 64 * 2);
  // packed weights
  auto packBytes = [](int K, int N) -> size_t {
    return (size_t)((K + 31) / 32) * ((N + 15) / 16) * 1024;
  };
  __bf16* wp_ne1 = (__bf16*)alloc(packBytes(92, 128));
  __bf16* wp_ne2 = (__bf16*)alloc(packBytes(128, 128));
  __bf16* wp_ee1 = (__bf16*)alloc(packBytes(3, 128));
  __bf16* wp_ee2 = (__bf16*)alloc(packBytes(128, 128));
  __bf16 *wp_pe1[3], *wp_pe2[3], *wp_pn1[3], *wp_pn2[3];
  for (int l = 0; l < 3; ++l) {
    wp_pe1[l] = (__bf16*)alloc(packBytes(384, 256));
    wp_pe2[l] = (__bf16*)alloc(packBytes(256, 128));
    wp_pn1[l] = (__bf16*)alloc(packBytes(256, 256));
    wp_pn2[l] = (__bf16*)alloc(packBytes(256, 128));
  }
  __bf16 *wp_q[2], *wp_k[2], *wp_v[2], *wp_o[2], *wp_f1[2], *wp_f2[2];
  for (int t = 0; t < 2; ++t) {
    wp_q[t]  = (__bf16*)alloc(packBytes(128, 128));
    wp_k[t]  = (__bf16*)alloc(packBytes(128, 128));
    wp_v[t]  = (__bf16*)alloc(packBytes(128, 128));
    wp_o[t]  = (__bf16*)alloc(packBytes(128, 128));
    wp_f1[t] = (__bf16*)alloc(packBytes(128, 512));
    wp_f2[t] = (__bf16*)alloc(packBytes(512, 128));
  }
  __bf16* wp_dec = (__bf16*)alloc(packBytes(128, 128));
  __bf16* wp_o1  = (__bf16*)alloc(packBytes(128, 64));
  __bf16* wp_o2  = (__bf16*)alloc(packBytes(64, 1));

  // --- launch helpers ---
  auto pack = [&](const float* W, __bf16* dst, int K, int N) {
    int frags = ((K + 31) / 32) * ((N + 15) / 16);
    k_pack_w<<<dim3((frags + 3) / 4), 128, 0, stream>>>(W, dst, K, N);
  };
  // single-source GEMM; N multiple of 64 -> k_gemm<4>, else k_gemm<1>
  auto gemm1 = [&](const __bf16* A, int Kt, int lda, const __bf16* Wp,
                   const float* bias, const float* resid, float* C, __bf16* Cb,
                   int M, int Nd, int act, const float* slope) {
    if (Nd % 64 == 0) {
      dim3 g((M + 127) / 128, Nd / 64, 1);
      k_gemm<4><<<g, 256, 0, stream>>>(A, nullptr, Kt, nullptr, nullptr, 0,
                                       nullptr, nullptr, 0, lda, Wp, bias, resid,
                                       C, Cb, M, Nd, Nd, act, slope);
    } else {
      dim3 g((M + 127) / 128, (Nd + 15) / 16, 1);
      k_gemm<1><<<g, 256, 0, stream>>>(A, nullptr, Kt, nullptr, nullptr, 0,
                                       nullptr, nullptr, 0, lda, Wp, bias, resid,
                                       C, Cb, M, Nd, Nd, act, slope);
    }
  };
  auto ln = [&](const float* in, float* outf, __bf16* outb, int rows, int D,
                const float* g_, const float* b_, const float* slope) {
    k_layernorm<<<dim3((rows + 7) / 8), 256, 0, stream>>>(in, outf, outb, rows, D,
                                                          g_, b_, slope);
  };

  // --- pack all weights (bf16 WMMA fragments) ---
  pack(ne_l1w, wp_ne1, 92, 128);  pack(ne_l2w, wp_ne2, 128, 128);
  pack(ee_l1w, wp_ee1, 3, 128);   pack(ee_l2w, wp_ee2, 128, 128);
  for (int l = 0; l < 3; ++l) {
    pack(pe_l1w[l], wp_pe1[l], 384, 256);
    pack(pe_l2w[l], wp_pe2[l], 256, 128);
    pack(pn_l1w[l], wp_pn1[l], 256, 256);
    pack(pn_l2w[l], wp_pn2[l], 256, 128);
  }
  for (int t = 0; t < 2; ++t) {
    pack(wqw[t], wp_q[t], 128, 128); pack(wkw[t], wp_k[t], 128, 128);
    pack(wvw[t], wp_v[t], 128, 128); pack(wow[t], wp_o[t], 128, 128);
    pack(f1w[t], wp_f1[t], 128, 512); pack(f2w[t], wp_f2[t], 512, 128);
  }
  pack(dec_w, wp_dec, 128, 128);
  pack(out_l1w, wp_o1, 128, 64);
  pack(out_l2w, wp_o2, 64, 1);

  // --- bf16 inputs ---
  k_cvt_pad<<<dim3((unsigned)(((long long)Nn * 96 + 255) / 256)), 256, 0, stream>>>(
      x_in, xb, Nn, 92, 96);
  k_edge_input<<<dim3((E + 255) / 256), 256, 0, stream>>>(evec, eshift, e_inb, E);

  // --- encoders ---
  gemm1(xb, 3, 96, wp_ne1, ne_l1b, nullptr, nullptr, tmpNb, Nn, 128, 1, ne_a);
  gemm1(tmpNb, 4, 128, wp_ne2, ne_l2b, nullptr, xh, xhb, Nn, 128, 0, nullptr);
  for (int c = 0; c < NCHUNK; ++c) {
    int c0 = c * EC;
    gemm1(e_inb + (size_t)c0 * 32, 1, 32, wp_ee1, ee_l1b, nullptr,
          nullptr, eb128b, EC, 128, 1, ee_a);
    gemm1(eb128b, 4, 128, wp_ee2, ee_l2b, nullptr,
          ea + (size_t)c0 * H, eab + (size_t)c0 * H, EC, 128, 0, nullptr);
  }

  // --- scatter-mean denominators (layer-invariant) ---
  k_fill<<<dim3((Nn + 255) / 256), 256, 0, stream>>>(cnt, Nn, 0.0f);
  k_count<<<dim3((E + 255) / 256), 256, 0, stream>>>(ei_col, cnt, E);

  // --- message passing (3 layers, edge pipeline chunked) ---
  for (int l = 0; l < 3; ++l) {
    k_fill<<<dim3((unsigned)(((long long)Nn * H + 255) / 256)), 256, 0, stream>>>(
        sums, (long long)Nn * H, 0.0f);
    for (int c = 0; c < NCHUNK; ++c) {
      int c0 = c * EC;
      dim3 g((EC + 127) / 128, 4, 1);  // Nd=256 -> 4 x 64-wide tiles
      k_gemm<4><<<g, 256, 0, stream>>>(xhb, ei_row + c0, 4,
                                       xhb, ei_col + c0, 4,
                                       eab + (size_t)c0 * H, nullptr, 4,
                                       128, wp_pe1[l], pe_l1b[l], nullptr,
                                       eb256, nullptr, EC, 256, 256, 0, nullptr);
      ln(eb256, nullptr, eb256b, EC, 256, pe_lng[l], pe_lnb[l], pe_a[l]);
      gemm1(eb256b, 8, 256, wp_pe2[l], pe_l2b[l], nullptr, eb128, nullptr,
            EC, 128, 0, nullptr);
      k_scatter<<<dim3((EC * H + 255) / 256), 256, 0, stream>>>(eb128, ei_col + c0,
                                                                sums, EC);
      k_add<<<dim3((EC * H + 255) / 256), 256, 0, stream>>>(
          ea + (size_t)c0 * H, eab + (size_t)c0 * H, eb128, EC * H);
    }
    k_aggdiv<<<dim3((unsigned)(((long long)Nn * H + 255) / 256)), 256, 0, stream>>>(
        aggb, sums, cnt, Nn * H);
    dim3 gn((Nn + 127) / 128, 4, 1);  // Nd=256
    k_gemm<4><<<gn, 256, 0, stream>>>(xhb, nullptr, 4, aggb, nullptr, 4,
                                      nullptr, nullptr, 0, 128, wp_pn1[l], pn_l1b[l],
                                      nullptr, nb256, nullptr, Nn, 256, 256, 0, nullptr);
    ln(nb256, nullptr, nb256b, Nn, 256, pn_lng[l], pn_lnb[l], pn_a[l]);
    gemm1(nb256b, 8, 256, wp_pn2[l], pn_l2b[l], xh, xh, xhb, Nn, 128, 0, nullptr);
  }

  // --- cross-attention transformer ---
  k_init_q<<<dim3((B * NE * H + 255) / 256), 256, 0, stream>>>(qb, emb, B * NE * H);
  const float scale = 0.088388347648318447f;  // 1/sqrt(128)
  for (int t = 0; t < 2; ++t) {
    ln(qb, nullptr, qnb, B * NE, 128, ln1g[t], ln1b[t], nullptr);
    ln(xh, nullptr, kvnb, Nn, 128, ln1g[t], ln1b[t], nullptr);
    gemm1(qnb, 4, 128, wp_q[t], wqb[t], nullptr, nullptr, Qmb, B * NE, 128, 0, nullptr);
    gemm1(kvnb, 4, 128, wp_k[t], wkb[t], nullptr, nullptr, Kmb, Nn, 128, 0, nullptr);
    gemm1(kvnb, 4, 128, wp_v[t], wvb[t], nullptr, Vm, nullptr, Nn, 128, 0, nullptr);
    k_gemm_nt<<<dim3(1, 5, B), 256, 0, stream>>>(Qmb, Kmb, sc, NE, P, 4, H, H, P,
                                                 (long long)NE * H, (long long)P * H,
                                                 (long long)NE * P, scale);
    k_softmax<<<dim3((B * NE + 7) / 8), 256, 0, stream>>>(sc, scb, B * NE, P);
    k_gemm_nn<<<dim3(1, 4, B), 256, 0, stream>>>(scb, Vm, ctxb, NE, H, 5, P, H, H,
                                                 (long long)NE * P, (long long)P * H,
                                                 (long long)NE * H);
    gemm1(ctxb, 4, 128, wp_o[t], wob[t], qb, qb, nullptr, B * NE, 128, 0, nullptr);
    ln(qb, nullptr, qnb, B * NE, 128, ln2g[t], ln2b[t], nullptr);
    gemm1(qnb, 4, 128, wp_f1[t], f1b[t], nullptr, nullptr, ffhb, B * NE, 512, 2, nullptr);
    gemm1(ffhb, 16, 512, wp_f2[t], f2b[t], qb, qb, nullptr, B * NE, 128, 0, nullptr);
  }
  ln(qb, en, nullptr, B * NE, 128, trln_g, trln_b, nullptr);

  // --- graph decoder + fused output head ---
  k_sum_nodes<<<dim3(B), 128, 0, stream>>>(xh, gsumb);
  gemm1(gsumb, 4, 128, wp_dec, dec_b, nullptr, graph, nullptr, B, 128, 0, nullptr);
  k_fused<<<dim3((B * NE * H + 255) / 256), 256, 0, stream>>>(fusedb, en, graph,
                                                              alpha, B * NE * H);
  gemm1(fusedb, 4, 128, wp_o1, out_l1b, nullptr, oh1, nullptr, B * NE, 64, 0, nullptr);
  ln(oh1, nullptr, oh1b, B * NE, 64, out_lng, out_lnb, out_a);
  gemm1(oh1b, 2, 64, wp_o2, out_l2b, nullptr, (float*)d_out, nullptr, B * NE, 1, 0, nullptr);
}
